// myECM_lstm_75797582840309
// MI455X (gfx1250) — compile-verified, matrix operand order checked by reading, not run on hardware
//
#include <hip/hip_runtime.h>
#include <hip/hip_bf16.h>
#include <cstdint>

// ---------------------------------------------------------------------------
// Problem constants (from reference): B=8192, T=21, F=3, U=512, 4U=2048
// ---------------------------------------------------------------------------
#define BB   8192
#define TT   21
#define FF   3
#define UU   512
#define U4   2048

typedef __attribute__((ext_vector_type(16))) __bf16 v16bf;
typedef __attribute__((ext_vector_type(8)))  float  v8f;

union Frag { uint4 q[2]; v16bf v; };

// ---------------------------------------------------------------------------
// Zero-fill (u32 words)
// ---------------------------------------------------------------------------
__global__ void zero_u32(uint32_t* p, size_t nwords) {
    size_t i = (size_t)blockIdx.x * blockDim.x + threadIdx.x;
    if (i < nwords) p[i] = 0u;
}

// ---------------------------------------------------------------------------
// Pack a row-major f32 B matrix [K,N] (optionally split into two K-sources)
// into bf16 WMMA B fragments.  Fragment f = nt*KT + kt covers K=kt*32..+31,
// N=nt*16..+15.  Per lane L: n = nt*16 + (L&15); kbase = kt*32 + (L>=16 ? 8:0);
// elems 0..7 -> kbase..kbase+7 ; elems 8..15 -> kbase+16..kbase+23.
// Launch: grid = NT*KT fragments, block = 32.
// ---------------------------------------------------------------------------
__global__ void pack_B(const float* __restrict__ s0, const float* __restrict__ s1,
                       int K0, int K, int N, __bf16* __restrict__ dst) {
    int f    = blockIdx.x;
    int lane = threadIdx.x;           // 0..31
    int KT   = K >> 5;
    int nt   = f / KT;
    int kt   = f - nt * KT;
    int n     = nt * 16 + (lane & 15);
    int kbase = kt * 32 + ((lane >> 4) << 3);
    __bf16* d = dst + (size_t)f * 512 + lane * 16;
    #pragma unroll
    for (int e = 0; e < 8; ++e) {
        int k = kbase + e;
        const float* s = (k < K0) ? s0 : s1;
        int kk = (k < K0) ? k : (k - K0);
        d[e] = (__bf16)s[(size_t)kk * N + n];
    }
    #pragma unroll
    for (int e = 0; e < 8; ++e) {
        int k = kbase + 16 + e;
        const float* s = (k < K0) ? s0 : s1;
        int kk = (k < K0) ? k : (k - K0);
        d[8 + e] = (__bf16)s[(size_t)kk * N + n];
    }
}

// ---------------------------------------------------------------------------
// Per-timestep mask forcing + batch-axis forward-fill gather index.
// learn: [T,B] int32.  One block per t; serial scan in LDS (cheap: 8K elems).
// ---------------------------------------------------------------------------
__global__ void mask_scan(const int* __restrict__ learn,
                          unsigned char* __restrict__ maskOut,
                          int* __restrict__ oidxOut) {
    __shared__ unsigned char sm[BB];
    __shared__ int slast[BB];
    __shared__ int sfirst;
    int t = blockIdx.x;
    const int* lt = learn + (size_t)t * BB;
    for (int b = threadIdx.x; b < BB; b += blockDim.x)
        sm[b] = (lt[b] == 1) ? 1 : 0;
    __syncthreads();
    if (threadIdx.x == 0) {
        int any = 0;
        for (int b = 0; b < BB; ++b) any |= sm[b];
        if (!any) sm[0] = 1;              // force row 0 to learn
        int run = -1, first = -1;
        for (int b = 0; b < BB; ++b) {
            if (sm[b]) { run = b; if (first < 0) first = b; }
            slast[b] = run;
        }
        sfirst = first;
    }
    __syncthreads();
    int first = sfirst;
    for (int b = threadIdx.x; b < BB; b += blockDim.x) {
        maskOut[(size_t)t * BB + b] = sm[b];
        int l = slast[b];
        oidxOut[(size_t)t * BB + b] = (l < 0) ? first : l;
    }
}

// ---------------------------------------------------------------------------
// z init for timestep t: z[b,n] = bias[n] + sum_f x[b,t,f]*kernel[f,n]
// ---------------------------------------------------------------------------
__global__ void init_z(const float* __restrict__ x, const float* __restrict__ kern,
                       const float* __restrict__ bias, float* __restrict__ z, int t) {
    size_t idx = (size_t)blockIdx.x * blockDim.x + threadIdx.x;   // over B*2048
    int b = (int)(idx >> 11);
    int n = (int)(idx & 2047);
    const float* xr = x + ((size_t)b * TT + t) * FF;
    z[idx] = bias[n] + xr[0] * kern[n] + xr[1] * kern[U4 + n] + xr[2] * kern[2 * U4 + n];
}

// ---------------------------------------------------------------------------
// bf16 WMMA GEMM:  C[M,N] = A @ B  with A bf16 row-major (row stride 512
// elements; K split across two sources at K0), B pre-packed fragments.
// Block = 128 threads = 4 waves (2 M x 2 N), wave tile = 64x64 (4x4 frags,
// 128 accumulator VGPRs), block tile = 128(M) x 128(N).
// K loop is split per A-source so the hot loop has no source selects; all
// addresses are incrementing per-lane pointers.
//   mode 0 (LSTM z):  zbuf[row*N+col] += acc
//   mode 1 (RNN)   :  hout[row*N+col] = bf16( tanh(acc + bias[col]) )
// ---------------------------------------------------------------------------
__global__ __launch_bounds__(128)
void gemm_wmma(const __bf16* __restrict__ A0, const __bf16* __restrict__ A1,
               int K0, int Ktot,
               const __bf16* __restrict__ Bp, int N,
               float* __restrict__ zbuf, const float* __restrict__ bias,
               __bf16* __restrict__ hout, int mode) {
    const int lane  = threadIdx.x & 31;
    const int wave  = threadIdx.x >> 5;
    const int waveM = wave >> 1;                      // 0..1
    const int waveN = wave & 1;                       // 0..1
    const int mBase = blockIdx.x * 128 + waveM * 64;
    const int nBase = blockIdx.y * 128 + waveN * 64;
    const int KT    = Ktot >> 5;
    const int KT0   = K0 >> 5;

    const int rowInLane = lane & 15;
    const int koffLane  = (lane >> 4) << 3;

    v8f acc[4][4] = {};

    // B fragment pointers: frag (ntile, kt) at Bp + (ntile*KT + kt)*512 + lane*16
    const __bf16* bP[4];
    #pragma unroll
    for (int ni = 0; ni < 4; ++ni)
        bP[ni] = Bp + (size_t)(((nBase >> 4) + ni) * KT) * 512 + lane * 16;

    const __bf16* aP[4];
    auto setA = [&](const __bf16* Asrc) {
        #pragma unroll
        for (int mi = 0; mi < 4; ++mi)
            aP[mi] = Asrc + (size_t)(mBase + mi * 16 + rowInLane) * UU + koffLane;
    };

    auto kstep = [&]() {
        Frag a[4], b[4];
        #pragma unroll
        for (int mi = 0; mi < 4; ++mi) {
            const uint4* p = (const uint4*)aP[mi];
            a[mi].q[0] = p[0];     // K elems [koff .. koff+7]
            a[mi].q[1] = p[2];     // K elems [koff+16 .. koff+23]
            aP[mi] += 32;
        }
        #pragma unroll
        for (int ni = 0; ni < 4; ++ni) {
            const uint4* p = (const uint4*)bP[ni];
            b[ni].q[0] = p[0];
            b[ni].q[1] = p[1];
            bP[ni] += 512;
        }
        #pragma unroll
        for (int mi = 0; mi < 4; ++mi)
            #pragma unroll
            for (int ni = 0; ni < 4; ++ni)
                acc[mi][ni] = __builtin_amdgcn_wmma_f32_16x16x32_bf16(
                    false, a[mi].v, false, b[ni].v,
                    (short)0, acc[mi][ni], false, false);
    };

    setA(A0);
    for (int kt = 0; kt < KT0; ++kt) kstep();
    if (KT > KT0) {
        setA(A1);
        for (int kt = KT0; kt < KT; ++kt) kstep();
    }

    // Epilogue — C layout: VGPR r holds M = r (lanes 0-15) / M = 8+r (16-31),
    // N = lane&15.
    #pragma unroll
    for (int mi = 0; mi < 4; ++mi) {
        #pragma unroll
        for (int ni = 0; ni < 4; ++ni) {
            int col   = nBase + ni * 16 + (lane & 15);
            int rbase = mBase + mi * 16 + ((lane >> 4) << 3);
            #pragma unroll
            for (int r = 0; r < 8; ++r) {
                int row = rbase + r;
                float v = acc[mi][ni][r];
                if (mode == 0) {
                    size_t idx = (size_t)row * N + col;
                    zbuf[idx] += v;
                } else {
                    float hv = tanhf(v + bias[col]);
                    hout[(size_t)row * N + col] = (__bf16)hv;
                }
            }
        }
    }
}

// ---------------------------------------------------------------------------
// LSTM gates + cell update for timestep t (mask-gated), store o and tanh(c).
// ---------------------------------------------------------------------------
__global__ void gates(const float* __restrict__ z, float* __restrict__ c,
                      const unsigned char* __restrict__ mask_t,
                      float* __restrict__ o_comp, float* __restrict__ tc) {
    size_t idx = (size_t)blockIdx.x * blockDim.x + threadIdx.x;   // over B*512
    int b = (int)(idx >> 9);
    int u = (int)(idx & 511);
    const float* zr = z + (size_t)b * U4;
    float zi = zr[u], zf = zr[UU + u], zg = zr[2 * UU + u], zo = zr[3 * UU + u];
    float ig = 1.0f / (1.0f + __expf(-zi));
    float fg = 1.0f / (1.0f + __expf(-zf));
    float gg = tanhf(zg);
    float og = 1.0f / (1.0f + __expf(-zo));
    float cv = c[idx];
    float cn = mask_t[b] ? (fg * cv + ig * gg) : cv;
    c[idx]      = cn;
    o_comp[idx] = og;
    tc[idx]     = tanhf(cn);
}

// ---------------------------------------------------------------------------
// h = o_comp[o_idx[b]] * tanh(c); write bf16 h (next-step A) and hs[t].
// ---------------------------------------------------------------------------
__global__ void gather_h(const float* __restrict__ o_comp, const float* __restrict__ tc,
                         const int* __restrict__ oidx_t,
                         __bf16* __restrict__ hbf, __bf16* __restrict__ hs_t) {
    size_t idx = (size_t)blockIdx.x * blockDim.x + threadIdx.x;   // over B*512
    int b = (int)(idx >> 9);
    int u = (int)(idx & 511);
    int src = oidx_t[b];
    float h = o_comp[(size_t)src * UU + u] * tc[idx];
    __bf16 hb = (__bf16)h;
    hbf[idx]  = hb;
    hs_t[idx] = hb;
}

// ---------------------------------------------------------------------------
// Head: out[b] = (h_last @ w_fc + b_fc) @ w_out + b_out.  One wave32 per row.
// ---------------------------------------------------------------------------
__global__ void fc_head(const __bf16* __restrict__ h, const float* __restrict__ w_fc,
                        const float* __restrict__ b_fc, const float* __restrict__ w_out,
                        const float* __restrict__ b_out, float* __restrict__ out) {
    int row  = blockIdx.x * (blockDim.x >> 5) + (threadIdx.x >> 5);
    int lane = threadIdx.x & 31;
    const __bf16* hr = h + (size_t)row * UU;
    float acc = 0.0f;
    for (int k = 0; k < UU; ++k)
        acc += (float)hr[k] * w_fc[(size_t)k * 32 + lane];
    float v = (acc + b_fc[lane]) * w_out[lane];
    #pragma unroll
    for (int off = 16; off > 0; off >>= 1)
        v += __shfl_down(v, off, 32);
    if (lane == 0) out[row] = v + b_out[0];
}

// ---------------------------------------------------------------------------
// Host launcher
// ---------------------------------------------------------------------------
static inline size_t alignUp(size_t v, size_t a) { return (v + a - 1) & ~(a - 1); }

extern "C" void kernel_launch(void* const* d_in, const int* in_sizes, int n_in,
                              void* d_out, int out_size, void* d_ws, size_t ws_size,
                              hipStream_t stream) {
    (void)in_sizes; (void)n_in; (void)out_size; (void)ws_size;
    const float* x      = (const float*)d_in[0];
    const int*   learn  = (const int*)  d_in[1];
    const float* kern   = (const float*)d_in[2];
    const float* rker   = (const float*)d_in[3];
    const float* bias   = (const float*)d_in[4];
    const float* w_rnn  = (const float*)d_in[5];
    const float* u_rnn  = (const float*)d_in[6];
    const float* b_rnn  = (const float*)d_in[7];
    const float* w_fc   = (const float*)d_in[8];
    const float* b_fc   = (const float*)d_in[9];
    const float* w_out  = (const float*)d_in[10];
    const float* b_out  = (const float*)d_in[11];
    float* out = (float*)d_out;

    // ---- workspace partition ----
    char* ws = (char*)d_ws;
    size_t off = 0;
    auto grab = [&](size_t bytes) { char* p = ws + off; off = alignUp(off + bytes, 256); return p; };
    __bf16* WrP   = (__bf16*)grab((size_t)(U4/16) * (UU/32)  * 1024);   // 2 MB
    __bf16* WcatP = (__bf16*)grab((size_t)(UU/16) * (2*UU/32) * 1024);  // 1 MB
    float*  zbuf  = (float*) grab((size_t)BB * U4 * 4);                 // 64 MB
    float*  cbuf  = (float*) grab((size_t)BB * UU * 4);                 // 16 MB
    float*  oco   = (float*) grab((size_t)BB * UU * 4);                 // 16 MB
    float*  tcb   = (float*) grab((size_t)BB * UU * 4);                 // 16 MB
    __bf16* hbf   = (__bf16*)grab((size_t)BB * UU * 2);                 // 8 MB
    __bf16* hrA   = (__bf16*)grab((size_t)BB * UU * 2);                 // 8 MB
    __bf16* hrB   = (__bf16*)grab((size_t)BB * UU * 2);                 // 8 MB
    __bf16* hs    = (__bf16*)grab((size_t)TT * BB * UU * 2);            // 176 MB
    unsigned char* maskB = (unsigned char*)grab((size_t)TT * BB);
    int*    oidx  = (int*)   grab((size_t)TT * BB * 4);

    // ---- weight packing ----
    {
        int fragsWr = (U4 / 16) * (UU / 32);                 // 128*16 = 2048
        pack_B<<<fragsWr, 32, 0, stream>>>(rker, rker, UU, UU, U4, WrP);
        int fragsWc = (UU / 16) * ((2 * UU) / 32);           // 32*32 = 1024
        pack_B<<<fragsWc, 32, 0, stream>>>(w_rnn, u_rnn, UU, 2 * UU, UU, WcatP);
    }

    // ---- mask / gather-index precompute ----
    mask_scan<<<TT, 256, 0, stream>>>(learn, maskB, oidx);

    // ---- zero init: h (bf16), c (f32), rnn h (bf16) ----
    {
        size_t w1 = (size_t)BB * UU * 2 / 4;   // bf16 buffers, words
        size_t w2 = (size_t)BB * UU;           // f32 buffer, words
        zero_u32<<<(int)((w1 + 255) / 256), 256, 0, stream>>>((uint32_t*)hbf, w1);
        zero_u32<<<(int)((w2 + 255) / 256), 256, 0, stream>>>((uint32_t*)cbuf, w2);
        zero_u32<<<(int)((w1 + 255) / 256), 256, 0, stream>>>((uint32_t*)hrA, w1);
    }

    // ---- LSTM scan over T ----
    const size_t elemsZ = (size_t)BB * U4;
    const size_t elemsH = (size_t)BB * UU;
    dim3 gemmGridZ(BB / 128, U4 / 128);
    for (int t = 0; t < TT; ++t) {
        init_z<<<(int)(elemsZ / 256), 256, 0, stream>>>(x, kern, bias, zbuf, t);
        gemm_wmma<<<gemmGridZ, 128, 0, stream>>>(hbf, hbf, UU, UU, WrP, U4,
                                                 zbuf, nullptr, nullptr, /*mode=*/0);
        gates<<<(int)(elemsH / 256), 256, 0, stream>>>(zbuf, cbuf,
                                                       maskB + (size_t)t * BB, oco, tcb);
        gather_h<<<(int)(elemsH / 256), 256, 0, stream>>>(oco, tcb,
                                                          oidx + (size_t)t * BB,
                                                          hbf, hs + (size_t)t * BB * UU);
    }

    // ---- SimpleRNN scan over T (ping-pong h) ----
    dim3 gemmGridR(BB / 128, UU / 128);
    __bf16* hcur = hrA;
    __bf16* hnxt = hrB;
    for (int t = 0; t < TT; ++t) {
        gemm_wmma<<<gemmGridR, 128, 0, stream>>>(hs + (size_t)t * BB * UU, hcur,
                                                 UU, 2 * UU, WcatP, UU,
                                                 nullptr, b_rnn, hnxt, /*mode=*/1);
        __bf16* tmp = hcur; hcur = hnxt; hnxt = tmp;
    }

    // ---- FC head: 8 rows per 256-thread block ----
    fc_head<<<BB / 8, 256, 0, stream>>>(hcur, w_fc, b_fc, w_out, b_out, out);
}
